// CosBahAttention_326417515270
// MI455X (gfx1250) — matrix operand order repaired and checked by reading, not compiled, not run on hardware
//
#include <hip/hip_runtime.h>
#include <hip/hip_bf16.h>
#include <math.h>

#define BDIM 32
#define SDIM 2048
#define DDIM 1024
#define COS_EPS 1e-8f
#define STR 1040          // padded LDS row stride (bf16 elems), 16B-aligned rows

typedef __attribute__((ext_vector_type(16))) __bf16 v16bf;
typedef __attribute__((ext_vector_type(8)))  __bf16 v8bf;
typedef __attribute__((ext_vector_type(4)))  __bf16 v4bf;
typedef __attribute__((ext_vector_type(8)))  float  v8f;

// ---------- helpers ----------
__device__ __forceinline__ float fast_tanhf(float x) {
#if __has_builtin(__builtin_amdgcn_tanhf)
    return __builtin_amdgcn_tanhf(x);            // v_tanh_f32 (gfx1250 trans op)
#elif __has_builtin(__builtin_amdgcn_tanh_f32)
    return __builtin_amdgcn_tanh_f32(x);
#else
    return tanhf(x);
#endif
}

__device__ __forceinline__ float red16(float v) {
    // sum across a 16-lane group (xor network stays within lanes [0..15] / [16..31])
    v += __shfl_xor(v, 1, 32);
    v += __shfl_xor(v, 2, 32);
    v += __shfl_xor(v, 4, 32);
    v += __shfl_xor(v, 8, 32);
    return v;
}

__device__ __forceinline__ float block_reduce_max(float v, float* red, int tid) {
    red[tid] = v; __syncthreads();
    for (int o = 128; o > 0; o >>= 1) {
        if (tid < o) red[tid] = fmaxf(red[tid], red[tid + o]);
        __syncthreads();
    }
    float r = red[0]; __syncthreads();
    return r;
}

__device__ __forceinline__ float block_reduce_sum(float v, float* red, int tid) {
    red[tid] = v; __syncthreads();
    for (int o = 128; o > 0; o >>= 1) {
        if (tid < o) red[tid] += red[tid + o];
        __syncthreads();
    }
    float r = red[0]; __syncthreads();
    return r;
}

// ---------- kernel 1a: Wb fp32 -> bf16 (L2-resident B matrix) ----------
__global__ __launch_bounds__(256) void cvt_wb_kernel(const float* __restrict__ Wb,
                                                     unsigned short* __restrict__ wbb) {
    int i = blockIdx.x * 256 + threadIdx.x;
    if (i < DDIM * DDIM) ((__bf16*)wbb)[i] = (__bf16)Wb[i];
}

// ---------- kernel 1b: k_n = k / max(||k||, eps) ----------
__global__ __launch_bounds__(256) void knorm_kernel(const float* __restrict__ k,
                                                    float* __restrict__ k_n) {
    __shared__ float red[256];
    int b = blockIdx.x, tid = threadIdx.x;
    float s = 0.f;
    for (int d = tid; d < DDIM; d += 256) { float v = k[b * DDIM + d]; s += v * v; }
    float nrm2 = block_reduce_sum(s, red, tid);
    float inv = 1.f / fmaxf(sqrtf(nrm2), COS_EPS);
    for (int d = tid; d < DDIM; d += 256) k_n[b * DDIM + d] = k[b * DDIM + d] * inv;
}

// ---------- kernel 1c: wk = k @ Wa.T  (32x1024x1024, tiny) ----------
__global__ __launch_bounds__(256) void wk_kernel(const float* __restrict__ k,
                                                 const float* __restrict__ Wa,
                                                 float* __restrict__ wk) {
    int e = blockIdx.x * 8 + (threadIdx.x >> 5);   // output column
    int b = threadIdx.x & 31;                      // batch row (one per lane)
    const float* kr = k + (long)b * DDIM;
    const float* wr = Wa + (long)e * DDIM;
    float acc = 0.f;
    for (int d = 0; d < DDIM; ++d) acc += kr[d] * wr[d];
    wk[(long)b * DDIM + e] = acc;
}

// ---------- kernel 2: fused score kernel (single pass over xs) ----------
// Per workgroup: 16 consecutive (b,s) rows. Stage xs tile (bf16) in LDS,
// compute cosine scores from the fp32 loads, then 8 waves x 8 n-tiles run the
// bf16 WMMA GEMM vs Wb, fusing +wk, tanh, *energy, column reduction.
__global__ __launch_bounds__(256) void score_kernel(const float* __restrict__ xs,
                                                    const float* __restrict__ k_n,
                                                    const float* __restrict__ wk,
                                                    const unsigned short* __restrict__ wbb_raw,
                                                    const float* __restrict__ energy,
                                                    float* __restrict__ sc_cos,
                                                    float* __restrict__ sc_bah) {
    __shared__ __bf16 tA[16 * STR];
    __shared__ float w_acc[16];
    const __bf16* wbb = (const __bf16*)wbb_raw;

    const int tid  = threadIdx.x;
    const int lane = tid & 31;
    const int wave = tid >> 5;
    const long tile = (long)blockIdx.x * 16;       // flat (b*S + s) row base
    const int b  = (int)(tile / SDIM);
    const int s0 = (int)(tile % SDIM);

    // ---- load + fp32->bf16 convert + row norms + k_n dot (cosine scores) ----
    {
        const int r = tid >> 4;                    // row 0..15 (aligned to 16-lane groups)
        const int c = tid & 15;
        const float* xrow  = xs  + (tile + r) * (long)DDIM;
        const float* knrow = k_n + (long)b * DDIM;
        float nrm = 0.f, dk = 0.f;
        #pragma unroll 4
        for (int j = 0; j < 16; ++j) {
            const int d0 = (j * 16 + c) * 4;
            const float4 x  = *(const float4*)(xrow + d0);
            const float4 kn = *(const float4*)(knrow + d0);
            nrm += x.x * x.x + x.y * x.y + x.z * x.z + x.w * x.w;
            dk  += kn.x * x.x + kn.y * x.y + kn.z * x.z + kn.w * x.w;
            v4bf bv;
            bv[0] = (__bf16)x.x; bv[1] = (__bf16)x.y;
            bv[2] = (__bf16)x.z; bv[3] = (__bf16)x.w;
            *(v4bf*)(&tA[r * STR + d0]) = bv;
        }
        nrm = red16(nrm);
        dk  = red16(dk);
        if (c == 0) sc_cos[(long)b * SDIM + s0 + r] = dk / fmaxf(sqrtf(nrm), COS_EPS);
    }
    if (tid < 16) w_acc[tid] = 0.f;
    __syncthreads();

    // ---- WMMA GEMM: wx = xs_tile @ Wb.T, fused tanh/energy reduction ----
    const int m    = lane & 15;                    // A row / B column within tile
    const int half = lane >> 4;
    float wpart[8];
    #pragma unroll
    for (int i = 0; i < 8; ++i) wpart[i] = 0.f;

    const __bf16* arow = tA + m * STR;
    for (int nt = wave * 8; nt < wave * 8 + 8; ++nt) {      // 8 n-tiles per wave, 64 total
        const int e = nt * 16 + m;                           // global output column
        const __bf16* bptr = wbb + (long)e * DDIM;
        v8f acc = {};
        #pragma unroll 4
        for (int kk = 0; kk < DDIM; kk += 32) {
            // A 16x32 bf16 per ISA layout: lane holds K = {kk+8*half..+7} U {kk+16+8*half..+7}
            v8bf c0 = *(const v8bf*)(arow + kk + 8 * half);
            v8bf c1 = *(const v8bf*)(arow + kk + 16 + 8 * half);
            v16bf afrag = (v16bf)__builtin_shufflevector(
                c0, c1, 0, 1, 2, 3, 4, 5, 6, 7, 8, 9, 10, 11, 12, 13, 14, 15);
            // B 32x16 bf16: lane = column e, same K order as A: {kk+8*half..+7} U {kk+16+8*half..+7}
            v8bf d0 = *(const v8bf*)(bptr + kk + 8 * half);
            v8bf d1 = *(const v8bf*)(bptr + kk + 16 + 8 * half);
            v16bf bfrag = (v16bf)__builtin_shufflevector(
                d0, d1, 0, 1, 2, 3, 4, 5, 6, 7, 8, 9, 10, 11, 12, 13, 14, 15);
            acc = __builtin_amdgcn_wmma_f32_16x16x32_bf16(
                false, afrag, false, bfrag, (short)0, acc, false, false);
        }
        // fuse: t = tanh(wx + wk[b,e]) * energy[e]; reduce over the 16 columns
        const float wkv = wk[(long)b * DDIM + nt * 16 + m];
        const float en  = energy[nt * 16 + m];
        #pragma unroll
        for (int i = 0; i < 8; ++i) {               // C/D layout: row = i + 8*half
            float t = fast_tanhf(acc[i] + wkv) * en;
            wpart[i] += red16(t);
        }
    }
    if (m == 0) {
        const int rbase = half * 8;
        #pragma unroll
        for (int i = 0; i < 8; ++i) atomicAdd(&w_acc[rbase + i], wpart[i]);  // ds_add_f32
    }
    __syncthreads();
    if (tid < 16) sc_bah[(long)b * SDIM + s0 + tid] = w_acc[tid];
}

// ---------- kernel 3: dual masked softmax + atts output + zero repr accum ----------
__global__ __launch_bounds__(256) void softmax_kernel(const float* __restrict__ sc_cos,
                                                      const float* __restrict__ sc_bah,
                                                      const unsigned char* __restrict__ mask,
                                                      float* __restrict__ a_cos,
                                                      float* __restrict__ a_bah,
                                                      float* __restrict__ atts,
                                                      float* __restrict__ cr,
                                                      float* __restrict__ br) {
    __shared__ float red[256];
    const int b = blockIdx.x, tid = threadIdx.x;
    // zero the repr accumulators (kernel 4 uses global atomics)
    for (int i = tid; i < DDIM; i += 256) { cr[b * DDIM + i] = 0.f; br[b * DDIM + i] = 0.f; }

    const unsigned char* mrow = mask + (long)b * SDIM;
    float vc[8], vb[8];
    float mxc = -INFINITY, mxb = -INFINITY;
    #pragma unroll
    for (int j = 0; j < 8; ++j) {
        const int s = tid + j * 256;
        const bool mm = mrow[s] != 0;
        vc[j] = mm ? sc_cos[(long)b * SDIM + s] : -INFINITY;
        vb[j] = mm ? sc_bah[(long)b * SDIM + s] : -INFINITY;
        mxc = fmaxf(mxc, vc[j]);
        mxb = fmaxf(mxb, vb[j]);
    }
    mxc = block_reduce_max(mxc, red, tid);
    mxb = block_reduce_max(mxb, red, tid);
    float sumc = 0.f, sumb = 0.f;
    float ec[8], eb[8];
    #pragma unroll
    for (int j = 0; j < 8; ++j) {
        ec[j] = __expf(vc[j] - mxc); sumc += ec[j];
        eb[j] = __expf(vb[j] - mxb); sumb += eb[j];
    }
    sumc = block_reduce_sum(sumc, red, tid);
    sumb = block_reduce_sum(sumb, red, tid);
    const float ic = 1.f / sumc, ib = 1.f / sumb;
    #pragma unroll
    for (int j = 0; j < 8; ++j) {
        const int s = tid + j * 256;
        const float pc = ec[j] * ic, pb = eb[j] * ib;
        a_cos[(long)b * SDIM + s] = pc;
        a_bah[(long)b * SDIM + s] = pb;
        atts[(long)b * SDIM + s]  = 0.5f * (pc + pb);
    }
}

// ---------- kernel 4: reprs = a @ xs (second streaming pass over xs) ----------
#define SSPLIT 8
__global__ __launch_bounds__(256) void repr_kernel(const float* __restrict__ xs,
                                                   const float* __restrict__ a_cos,
                                                   const float* __restrict__ a_bah,
                                                   float* __restrict__ cr,
                                                   float* __restrict__ br) {
    const int bi = blockIdx.x;
    const int b = bi / (4 * SSPLIT);
    const int dchunk = (bi / SSPLIT) & 3;
    const int sc = bi % SSPLIT;
    const int d = dchunk * 256 + threadIdx.x;
    const int sbeg = sc * (SDIM / SSPLIT), send = sbeg + SDIM / SSPLIT;
    float ac = 0.f, ab = 0.f;
    for (int s = sbeg; s < send; ++s) {
        const float pc = a_cos[(long)b * SDIM + s];
        const float pb = a_bah[(long)b * SDIM + s];
        const float x  = xs[((long)b * SDIM + s) * DDIM + d];
        __builtin_prefetch(&xs[((long)b * SDIM + s + 4) * DDIM + d], 0, 1);  // global_prefetch_b8
        ac += pc * x;
        ab += pb * x;
    }
    atomicAdd(&cr[(long)b * DDIM + d], ac);
    atomicAdd(&br[(long)b * DDIM + d], ab);
}

// ---------- kernel 5: attn = [cos_repr ; bah_repr] @ Wc.T + bc ----------
__global__ __launch_bounds__(256) void final_kernel(const float* __restrict__ cr,
                                                    const float* __restrict__ br,
                                                    const float* __restrict__ Wc,
                                                    const float* __restrict__ bc,
                                                    float* __restrict__ attn) {
    const int e = blockIdx.x * 8 + (threadIdx.x >> 5);
    const int b = threadIdx.x & 31;
    const float* w0 = Wc + (long)e * (2 * DDIM);
    const float* w1 = w0 + DDIM;
    const float* c  = cr + (long)b * DDIM;
    const float* bb = br + (long)b * DDIM;
    float acc = 0.f;
    for (int d = 0; d < DDIM; ++d) acc += c[d] * w0[d] + bb[d] * w1[d];
    attn[(long)b * DDIM + e] = acc + bc[e];
}

// ---------- host ----------
extern "C" void kernel_launch(void* const* d_in, const int* in_sizes, int n_in,
                              void* d_out, int out_size, void* d_ws, size_t ws_size,
                              hipStream_t stream) {
    (void)in_sizes; (void)n_in; (void)out_size; (void)ws_size;
    const float* k      = (const float*)d_in[0];
    const float* xs     = (const float*)d_in[1];
    const unsigned char* mask = (const unsigned char*)d_in[2];
    const float* Wa     = (const float*)d_in[3];
    const float* Wb     = (const float*)d_in[4];
    const float* energy = (const float*)d_in[5];
    const float* Wc     = (const float*)d_in[6];
    const float* bc     = (const float*)d_in[7];

    float* attn = (float*)d_out;                       // [B, D]
    float* atts = (float*)d_out + BDIM * DDIM;         // [B, S]

    // workspace carve (256B aligned)
    char* ws = (char*)d_ws;
    size_t off = 0;
    auto carve = [&](size_t bytes) { char* p = ws + off; off += (bytes + 255) & ~(size_t)255; return p; };
    unsigned short* wbb = (unsigned short*)carve((size_t)DDIM * DDIM * 2);  // Wb in bf16
    float* k_n    = (float*)carve((size_t)BDIM * DDIM * 4);
    float* wk     = (float*)carve((size_t)BDIM * DDIM * 4);
    float* sc_cos = (float*)carve((size_t)BDIM * SDIM * 4);
    float* sc_bah = (float*)carve((size_t)BDIM * SDIM * 4);
    float* a_cos  = (float*)carve((size_t)BDIM * SDIM * 4);
    float* a_bah  = (float*)carve((size_t)BDIM * SDIM * 4);
    float* cr     = (float*)carve((size_t)BDIM * DDIM * 4);
    float* br     = (float*)carve((size_t)BDIM * DDIM * 4);

    cvt_wb_kernel<<<(DDIM * DDIM + 255) / 256, 256, 0, stream>>>(Wb, wbb);
    knorm_kernel<<<BDIM, 256, 0, stream>>>(k, k_n);
    wk_kernel<<<DDIM / 8, 256, 0, stream>>>(k, Wa, wk);
    score_kernel<<<(BDIM * SDIM) / 16, 256, 0, stream>>>(xs, k_n, wk, wbb, energy, sc_cos, sc_bah);
    softmax_kernel<<<BDIM, 256, 0, stream>>>(sc_cos, sc_bah, mask, a_cos, a_bah, atts, cr, br);
    repr_kernel<<<BDIM * 4 * SSPLIT, 256, 0, stream>>>(xs, a_cos, a_bah, cr, br);
    final_kernel<<<DDIM / 8, 256, 0, stream>>>(cr, br, Wc, bc, attn);
}
